// GAT_46823733461096
// MI455X (gfx1250) — matrix operand order repaired and checked by reading, not compiled, not run on hardware
//
#include <hip/hip_runtime.h>
#include <hip/hip_bf16.h>

#define FIN   256
#define CH    128
#define NCLS  40
#define NEG_SLOPE 0.2f
#define BN_EPS 1e-5f

#define APAD  4              // TDM pad: 4 DWORDs per 256-DWORD row
#define AROW  (FIN + APAD)   // 260 floats per LDS row

typedef __attribute__((ext_vector_type(16))) __bf16       v16bf;
typedef __attribute__((ext_vector_type(8)))  float        v8f;
typedef __attribute__((ext_vector_type(2)))  float        v2f;
typedef __attribute__((ext_vector_type(4)))  unsigned int v4u;
typedef __attribute__((ext_vector_type(8)))  int          v8i;
typedef __attribute__((ext_vector_type(4)))  int          v4i;

// ---------------------------------------------------------------- utilities
__global__ void fill_kernel(float* __restrict__ p, float v, int n) {
    int i = blockIdx.x * blockDim.x + threadIdx.x;
    for (; i < n; i += gridDim.x * blockDim.x) p[i] = v;
}

__device__ __forceinline__ void atomicMaxF(float* addr, float v) {
    // ordering trick: works for mixed signs given -inf init
    if (v >= 0.0f) atomicMax((int*)addr, __float_as_int(v));
    else           atomicMin((unsigned int*)addr, __float_as_uint(v));
}

// ------------------------------------------------- GEMM1: h = x @ W  (bf16 WMMA)
// one block = one 16-row m-tile; A-tile DMA'd to LDS by the Tensor Data Mover,
// 8 waves each produce one 16x16 output tile across the 128 channels.
__global__ void gemm1_wmma_bf16(const float* __restrict__ x,
                                const float* __restrict__ W,
                                float* __restrict__ h, int n16) {
    __shared__ float smem[16 * AROW];           // 16.6 KB padded A tile
    const int mt   = blockIdx.x;
    const int nt   = threadIdx.x >> 5;          // 8 channel tiles
    const int lane = threadIdx.x & 31;

    if (threadIdx.x < 32) {
        // ---- TDM descriptor (D#): 16x256 f32 tile, row-padded into LDS ----
        unsigned lds_base = (unsigned)(size_t)&smem[0];
        unsigned long long ga =
            (unsigned long long)(size_t)(x + (size_t)mt * 16 * FIN);
        v4u g0 = {0u, 0u, 0u, 0u};
        g0[0] = 1u;                                     // count=1 (valid user D#)
        g0[1] = lds_base;                               // lds_addr
        g0[2] = (unsigned)(ga & 0xFFFFFFFFull);         // global_addr[31:0]
        g0[3] = (unsigned)((ga >> 32) & 0x1FFFFFFull)   // global_addr[56:32]
              | 0x80000000u;                            // type = 2 ("image")
        v8i g1 = {0, 0, 0, 0, 0, 0, 0, 0};
        g1[0] = (2 << 16)                               // data_size = 4 bytes
              | (1 << 20)                               // pad_enable
              | (7 << 22)                               // pad_interval = 256 DWORDs
              | (3 << 25);                              // pad_amount   = 4 DWORDs
        g1[1] = (FIN & 0xFFFF) << 16;                   // tensor_dim0 = 256 (lo16)
        g1[2] = (16 << 16);                             // dim0 hi16 | tensor_dim1 = 16
        g1[3] = (FIN << 16);                            // dim1 hi16 | tile_dim0 = 256
        g1[4] = 16;                                     // tile_dim1 = 16 | tile_dim2 = 0
        g1[5] = FIN;                                    // tensor_dim0_stride = 256
        v4i gz4 = {0, 0, 0, 0};
        v8i gz8 = {0, 0, 0, 0, 0, 0, 0, 0};
        __builtin_amdgcn_tensor_load_to_lds(g0, g1, gz4, gz4, gz8, 0);
        __builtin_amdgcn_s_wait_tensorcnt(0);           // issuing wave waits for TDM
    }
    __syncthreads();

    const int kb   = (lane & 16) ? 8 : 0;               // A: lane-half K split
    const int ncol = nt * 16;
    const int arow = (lane & 15) * AROW;                // A: lane = M (LDS row)

    v8f c = {};
    for (int k0 = 0; k0 < FIN; k0 += 32) {
        // A fragment from LDS: 16-bit A 16x32 layout (ISA 7.12.2)
        const float4* av0 = (const float4*)(&smem[arow + k0 + kb]);
        const float4* av1 = (const float4*)(&smem[arow + k0 + 16 + kb]);
        float4 p0 = av0[0], p1 = av0[1];
        float4 q0 = av1[0], q1 = av1[1];
        v16bf a;
        a[0]=(__bf16)p0.x; a[1]=(__bf16)p0.y; a[2]=(__bf16)p0.z; a[3]=(__bf16)p0.w;
        a[4]=(__bf16)p1.x; a[5]=(__bf16)p1.y; a[6]=(__bf16)p1.z; a[7]=(__bf16)p1.w;
        a[8]=(__bf16)q0.x; a[9]=(__bf16)q0.y; a[10]=(__bf16)q0.z; a[11]=(__bf16)q0.w;
        a[12]=(__bf16)q1.x; a[13]=(__bf16)q1.y; a[14]=(__bf16)q1.z; a[15]=(__bf16)q1.w;

        // B fragment from global (L2-resident 128 KB): lane = N, element = K (table order)
        const float* bp = W + (size_t)(k0 + kb) * CH + ncol + (lane & 15);
        float4 b0 = make_float4(bp[0 * CH], bp[1 * CH], bp[2 * CH], bp[3 * CH]);
        float4 b1 = make_float4(bp[4 * CH], bp[5 * CH], bp[6 * CH], bp[7 * CH]);
        float4 b2 = make_float4(bp[16 * CH], bp[17 * CH], bp[18 * CH], bp[19 * CH]);
        float4 b3 = make_float4(bp[20 * CH], bp[21 * CH], bp[22 * CH], bp[23 * CH]);
        v16bf b;
        b[0]=(__bf16)b0.x; b[1]=(__bf16)b0.y; b[2]=(__bf16)b0.z; b[3]=(__bf16)b0.w;
        b[4]=(__bf16)b1.x; b[5]=(__bf16)b1.y; b[6]=(__bf16)b1.z; b[7]=(__bf16)b1.w;
        b[8]=(__bf16)b2.x; b[9]=(__bf16)b2.y; b[10]=(__bf16)b2.z; b[11]=(__bf16)b2.w;
        b[12]=(__bf16)b3.x; b[13]=(__bf16)b3.y; b[14]=(__bf16)b3.z; b[15]=(__bf16)b3.w;

        c = __builtin_amdgcn_wmma_f32_16x16x32_bf16(
                false, a, false, b, (short)0, c, false, false);
    }

    // C/D layout: lane = N, VGPR r = M row (+8 for upper lane half)
    const int mb = mt * 16 + ((lane & 16) ? 8 : 0);
    const int nc = ncol + (lane & 15);
#pragma unroll
    for (int r = 0; r < 8; ++r)
        h[(size_t)(mb + r) * CH + nc] = c[r];
}

// ------------------------------------------------- attention scores per node
__global__ void attn_scores(const float* __restrict__ h,
                            const float* __restrict__ att_src,
                            const float* __restrict__ att_dst,
                            float* __restrict__ as_, float* __restrict__ ad_,
                            int n) {
    const int wave = blockIdx.x * 8 + (threadIdx.x >> 5);
    const int lane = threadIdx.x & 31;
    if (wave >= n) return;
    const float4 hv = *(const float4*)(h + (size_t)wave * CH + lane * 4);
    const float4 sv = *(const float4*)(att_src + lane * 4);
    const float4 dv = *(const float4*)(att_dst + lane * 4);
    float ps = hv.x*sv.x + hv.y*sv.y + hv.z*sv.z + hv.w*sv.w;
    float pd = hv.x*dv.x + hv.y*dv.y + hv.z*dv.z + hv.w*dv.w;
    for (int off = 16; off; off >>= 1) {
        ps += __shfl_xor(ps, off);
        pd += __shfl_xor(pd, off);
    }
    if (lane == 0) { as_[wave] = ps; ad_[wave] = pd; }
}

// ------------------------------------------------- edge pass A: leaky-relu logits + seg-max
__global__ void edge_logits(const int* __restrict__ ei, int E, int EN,
                            const float* __restrict__ as_, const float* __restrict__ ad_,
                            float* __restrict__ raw, float* __restrict__ amax) {
    int i = blockIdx.x * blockDim.x + threadIdx.x;
    if (i >= EN) return;
    int s, d;
    if (i < E) { s = ei[i]; d = ei[E + i]; } else { s = d = i - E; }
    float a = as_[s] + ad_[d];
    a = (a > 0.0f) ? a : NEG_SLOPE * a;
    raw[i] = a;
    atomicMaxF(amax + d, a);
}

// ------------------------------------------------- edge pass B: exp + seg-sum
__global__ void edge_exp(const int* __restrict__ ei, int E, int EN,
                         const float* __restrict__ amax,
                         float* __restrict__ eexp, float* __restrict__ asum) {
    int i = blockIdx.x * blockDim.x + threadIdx.x;
    if (i >= EN) return;
    int d = (i < E) ? ei[E + i] : (i - E);
    float e = __expf(eexp[i] - amax[d]);
    eexp[i] = e;
    atomicAdd(asum + d, e);
}

// ------------------------------------------------- edge pass C: weighted scatter-add
// one wave per edge per iteration; prefetch next edge's h row (global_prefetch)
__global__ void edge_aggregate(const int* __restrict__ ei, int E, int EN,
                               const float* __restrict__ h,
                               const float* __restrict__ eexp,
                               const float* __restrict__ asum,
                               float* __restrict__ agg) {
    const int lane   = threadIdx.x & 31;
    const int nwaves = gridDim.x * (blockDim.x >> 5);
    int e = blockIdx.x * (blockDim.x >> 5) + (threadIdx.x >> 5);
    for (; e < EN; e += nwaves) {
        int s, d;
        if (e < E) { s = ei[e]; d = ei[E + e]; } else { s = d = e - E; }
        int en = e + nwaves;
        if (en < EN) {
            int sn = (en < E) ? ei[en] : (en - E);
            __builtin_prefetch(h + (size_t)sn * CH + lane * 4, 0, 1);
        }
        float coef = 0.0f;
        if (lane == 0) coef = eexp[e] / (asum[d] + 1e-16f);
        coef = __shfl(coef, 0);
        const float4 hv = *(const float4*)(h + (size_t)s * CH + lane * 4);
        float* dst = agg + (size_t)d * CH + lane * 4;
        atomicAdd(dst + 0, hv.x * coef);
        atomicAdd(dst + 1, hv.y * coef);
        atomicAdd(dst + 2, hv.z * coef);
        atomicAdd(dst + 3, hv.w * coef);
    }
}

// ------------------------------------------------- bias + ReLU in place + BN partial sums
__global__ void bn_stats(float* __restrict__ agg, const float* __restrict__ bias,
                         float* __restrict__ bnsum, float* __restrict__ bnss, int n) {
    __shared__ float ls[256], lss[256];
    const int ch  = threadIdx.x & 127;
    const int sub = threadIdx.x >> 7;      // 0..1
    const float b = bias[ch];
    float s = 0.0f, ss = 0.0f;
    for (int r = blockIdx.x * 2 + sub; r < n; r += gridDim.x * 2) {
        float v = agg[(size_t)r * CH + ch] + b;
        v = fmaxf(v, 0.0f);
        agg[(size_t)r * CH + ch] = v;
        s += v; ss += v * v;
    }
    ls[threadIdx.x] = s; lss[threadIdx.x] = ss;
    __syncthreads();
    if (threadIdx.x < 128) {
        atomicAdd(bnsum + ch, ls[threadIdx.x] + ls[threadIdx.x + 128]);
        atomicAdd(bnss  + ch, lss[threadIdx.x] + lss[threadIdx.x + 128]);
    }
}

// ------------------------------------------------- fold BN affine into FC weights/bias
__global__ void bn_fc_prep(const float* __restrict__ bnsum, const float* __restrict__ bnss,
                           const float* __restrict__ gamma, const float* __restrict__ beta,
                           const float* __restrict__ fc_w, const float* __restrict__ fc_b,
                           float* __restrict__ fcwp, float* __restrict__ fcbp, int n) {
    __shared__ float sh_shift[128];
    const int t = threadIdx.x;                 // 0..127
    const float inv_n = 1.0f / (float)n;
    float mean = bnsum[t] * inv_n;
    float var  = bnss[t] * inv_n - mean * mean;
    float rsg  = rsqrtf(var + BN_EPS) * gamma[t];
    sh_shift[t] = beta[t] - mean * rsg;
    for (int c = 0; c < 48; ++c)
        fcwp[t * 48 + c] = (c < NCLS ? fc_w[t * NCLS + c] : 0.0f) * rsg;
    __syncthreads();
    if (t < 48) {
        float b = (t < NCLS) ? fc_b[t] : 0.0f;
        for (int k = 0; k < 128; ++k)
            b += sh_shift[k] * (t < NCLS ? fc_w[k * NCLS + t] : 0.0f);
        fcbp[t] = b;
    }
}

// ------------------------------------------------- FC: out = relu_out @ fcwp + fcbp (f32 WMMA)
__global__ void fc_wmma_f32(const float* __restrict__ agg,
                            const float* __restrict__ fcwp,   // [128][48]
                            const float* __restrict__ fcbp,   // [48]
                            float* __restrict__ out, int n16) {
    const int wave = blockIdx.x * 8 + (threadIdx.x >> 5);
    const int lane = threadIdx.x & 31;
    const int mt = wave / 3;
    const int nt = wave % 3;                   // 48 padded classes / 16
    if (mt >= n16) return;
    const int row = mt * 16 + (lane & 15);     // A: lane = M
    const int ko  = (lane & 16) ? 2 : 0;       // K split across lane halves
    const int nc  = nt * 16 + (lane & 15);     // B/D: lane = N
    v8f c = {};
    for (int k0 = 0; k0 < CH; k0 += 4) {
        v2f a, b;
        const float* ap = agg + (size_t)row * CH + k0 + ko;
        a[0] = ap[0]; a[1] = ap[1];
        const float* bp = fcwp + (size_t)(k0 + ko) * 48 + nc;
        b[0] = bp[0]; b[1] = bp[48];
        c = __builtin_amdgcn_wmma_f32_16x16x4_f32(
                false, a, false, b, (short)0, c, false, false);
    }
    if (nc < NCLS) {
        const float bb = fcbp[nc];
        const int mb = mt * 16 + ((lane & 16) ? 8 : 0);
#pragma unroll
        for (int r = 0; r < 8; ++r)
            out[(size_t)(mb + r) * NCLS + nc] = c[r] + bb;
    }
}

// ---------------------------------------------------------------- launcher
extern "C" void kernel_launch(void* const* d_in, const int* in_sizes, int n_in,
                              void* d_out, int out_size, void* d_ws, size_t ws_size,
                              hipStream_t stream) {
    const float* x        = (const float*)d_in[0];
    const float* W        = (const float*)d_in[1];
    const float* att_src  = (const float*)d_in[2];
    const float* att_dst  = (const float*)d_in[3];
    const float* bias_cv  = (const float*)d_in[4];
    const float* bn_gamma = (const float*)d_in[5];
    const float* bn_beta  = (const float*)d_in[6];
    const float* fc_w     = (const float*)d_in[7];
    const float* fc_b     = (const float*)d_in[8];
    const int*   ei       = (const int*)d_in[9];

    const int N  = in_sizes[0] / FIN;     // 100000
    const int E  = in_sizes[9] / 2;       // 1600000
    const int EN = E + N;
    const int N16 = N / 16;               // 6250 (exact)

    float* ws    = (float*)d_ws;
    float* h     = ws;                                  // N*128
    float* agg   = h     + (size_t)N * CH;              // N*128
    float* a_s   = agg   + (size_t)N * CH;              // N
    float* a_d   = a_s   + N;                           // N
    float* amax  = a_d   + N;                           // N
    float* asum  = amax  + N;                           // N
    float* eexp  = asum  + N;                           // E+N
    float* bnsum = eexp  + EN;                          // 128
    float* bnss  = bnsum + CH;                          // 128
    float* fcwp  = bnss  + CH;                          // 128*48
    float* fcbp  = fcwp  + CH * 48;                     // 48

    float ninf;
    { unsigned int u = 0xFF800000u; __builtin_memcpy(&ninf, &u, 4); }

    // zero/init accumulators (ws is poisoned by harness)
    fill_kernel<<<2048, 256, 0, stream>>>(agg,  0.0f, N * CH);
    fill_kernel<<<64,   256, 0, stream>>>(asum, 0.0f, N);
    fill_kernel<<<64,   256, 0, stream>>>(amax, ninf, N);
    fill_kernel<<<1,    256, 0, stream>>>(bnsum, 0.0f, 2 * CH);

    // h = x @ W  (TDM A-tile stage + bf16 WMMA, fp32 accumulate)
    gemm1_wmma_bf16<<<N16, 256, 0, stream>>>(x, W, h, N16);

    attn_scores<<<(N + 7) / 8, 256, 0, stream>>>(h, att_src, att_dst, a_s, a_d, N);
    edge_logits<<<(EN + 255) / 256, 256, 0, stream>>>(ei, E, EN, a_s, a_d, eexp, amax);
    edge_exp<<<(EN + 255) / 256, 256, 0, stream>>>(ei, E, EN, amax, eexp, asum);
    edge_aggregate<<<32768, 256, 0, stream>>>(ei, E, EN, h, eexp, asum, agg);
    bn_stats<<<512, 256, 0, stream>>>(agg, bias_cv, bnsum, bnss, N);
    bn_fc_prep<<<1, 128, 0, stream>>>(bnsum, bnss, bn_gamma, bn_beta, fc_w, fc_b, fcwp, fcbp, N);
    {
        int waves = N16 * 3;
        fc_wmma_f32<<<(waves + 7) / 8, 256, 0, stream>>>(agg, fcwp, fcbp, (float*)d_out, N16);
    }
}
